// LSABlock_77283641524902
// MI455X (gfx1250) — matrix operand order repaired and checked by reading, not compiled, not run on hardware
//
#include <hip/hip_runtime.h>
#include <hip/hip_bf16.h>
#include <math.h>

typedef __bf16 bf16_t;
typedef __attribute__((ext_vector_type(16))) __bf16 v16bf;
typedef __attribute__((ext_vector_type(8)))  __bf16 v8bf;
typedef __attribute__((ext_vector_type(8)))  float  v8f;

#define WMMA_BF16(a, b, c) \
  __builtin_amdgcn_wmma_f32_16x16x32_bf16(false, (a), false, (b), (short)0, (c), false, false)

__device__ __forceinline__ bf16_t f2bf(float f) {
  union { float f; unsigned int u; } v; v.f = f;
  unsigned int r = v.u + 0x7FFFu + ((v.u >> 16) & 1u);
  unsigned short h = (unsigned short)(r >> 16);
  bf16_t o; __builtin_memcpy(&o, &h, 2); return o;
}

__device__ __forceinline__ v16bf cat16(v8bf lo, v8bf hi) {
  return __builtin_shufflevector(lo, hi, 0,1,2,3,4,5,6,7,8,9,10,11,12,13,14,15);
}

// A fragment (16x32, MxK): lane r in 0..15 holds row r; lane group g=lane>>4
// holds K = g*8+{0..7} (elements 0..7) and 16+g*8+{0..7} (elements 8..15).
__device__ __forceinline__ v16bf load_frag_a(const bf16_t* base, int ld) {
  int lane = threadIdx.x & 31;
  int g = lane >> 4, r = lane & 15;
  const bf16_t* p = base + (size_t)r * ld + g * 8;
  v8bf lo = *(const v8bf*)p;
  v8bf hi = *(const v8bf*)(p + 16);
  return cat16(lo, hi);
}

// B fragment (32x16, KxN) sourced from row-major W (N rows of K): lane r holds
// column n=r; K = g*8+{0..7} (elements 0..7) and 16+g*8+{0..7} (elements 8..15).
__device__ __forceinline__ v16bf load_frag_b(const bf16_t* base, int ld) {
  int lane = threadIdx.x & 31;
  int g = lane >> 4, r = lane & 15;
  const bf16_t* p = base + (size_t)r * ld + g * 8;
  v8bf lo = *(const v8bf*)p;
  v8bf hi = *(const v8bf*)(p + 16);
  return cat16(lo, hi);
}

// ---------------------------------------------------------------------------
// fp32 -> bf16 cast (weights)
// ---------------------------------------------------------------------------
__global__ void cast_kernel(const float* __restrict__ in, bf16_t* __restrict__ out, int n) {
  int i = blockIdx.x * 256 + threadIdx.x;
  if (i < n) out[i] = f2bf(in[i]);
}

// ---------------------------------------------------------------------------
// LayerNorm over C=256, one block (256 threads) per row, bf16 output
// ---------------------------------------------------------------------------
__launch_bounds__(256)
__global__ void ln_kernel(const float* __restrict__ x, const float* __restrict__ gw,
                          const float* __restrict__ bw, bf16_t* __restrict__ out) {
  const int C = 256;
  __shared__ float red[8];
  int row = blockIdx.x;
  int t = threadIdx.x;
  int w = t >> 5, lane = t & 31;

  float v = x[(size_t)row * C + t];
  float s = v;
  #pragma unroll
  for (int m = 1; m < 32; m <<= 1) s += __shfl_xor(s, m, 32);
  if (lane == 0) red[w] = s;
  __syncthreads();
  float mean = 0.f;
  #pragma unroll
  for (int i = 0; i < 8; ++i) mean += red[i];
  mean *= (1.0f / C);

  float d = v - mean;
  float s2 = d * d;
  #pragma unroll
  for (int m = 1; m < 32; m <<= 1) s2 += __shfl_xor(s2, m, 32);
  __syncthreads();
  if (lane == 0) red[w] = s2;
  __syncthreads();
  float var = 0.f;
  #pragma unroll
  for (int i = 0; i < 8; ++i) var += red[i];
  var *= (1.0f / C);

  float rs = rsqrtf(var + 1e-5f);
  out[(size_t)row * C + t] = f2bf(d * rs * gw[t] + bw[t]);
}

// ---------------------------------------------------------------------------
// WMMA GEMM: out[m,o] = sum_k A[m,k] * W[o,k]  (+bias)(+GELU)(+residual)
// WG = 256 threads = 8 waves; WG tile 128x64; wave tile 32x32.
// ---------------------------------------------------------------------------
__launch_bounds__(256)
__global__ void gemm_wmma_kernel(const bf16_t* __restrict__ A, const bf16_t* __restrict__ W,
                                 int K, int O,
                                 const float* __restrict__ bias,
                                 const float* __restrict__ resid,
                                 bf16_t* __restrict__ out_bf,
                                 float* __restrict__ out_f,
                                 int act) {
  int w = threadIdx.x >> 5;
  int lane = threadIdx.x & 31;
  int wr = w & 3, wc = w >> 2;
  int row0 = blockIdx.x * 128 + wr * 32;
  int col0 = blockIdx.y * 64 + wc * 32;

  v8f c00{}, c01{}, c10{}, c11{};
  for (int k = 0; k < K; k += 32) {
    __builtin_prefetch(A + (size_t)row0 * K + k + 64, 0, 1);
    __builtin_prefetch(W + (size_t)col0 * K + k + 64, 0, 1);
    v16bf a0 = load_frag_a(A + (size_t)row0 * K + k, K);
    v16bf a1 = load_frag_a(A + (size_t)(row0 + 16) * K + k, K);
    v16bf b0 = load_frag_b(W + (size_t)col0 * K + k, K);
    v16bf b1 = load_frag_b(W + (size_t)(col0 + 16) * K + k, K);
    c00 = WMMA_BF16(a0, b0, c00);
    c01 = WMMA_BF16(a0, b1, c01);
    c10 = WMMA_BF16(a1, b0, c10);
    c11 = WMMA_BF16(a1, b1, c11);
  }

  int g = lane >> 4, r = lane & 15;
  v8f accs[4] = {c00, c01, c10, c11};
  #pragma unroll
  for (int ti = 0; ti < 2; ++ti) {
    #pragma unroll
    for (int tj = 0; tj < 2; ++tj) {
      v8f acc = accs[ti * 2 + tj];
      int col = col0 + tj * 16 + r;
      float bv = bias ? bias[col] : 0.f;
      #pragma unroll
      for (int j = 0; j < 8; ++j) {
        int row = row0 + ti * 16 + 8 * g + j;
        float val = acc[j] + bv;
        if (act == 1) val = 0.5f * val * (1.0f + erff(val * 0.70710678118654752f));
        if (resid) val += resid[(size_t)row * O + col];
        if (out_f) out_f[(size_t)row * O + col] = val;
        else       out_bf[(size_t)row * O + col] = f2bf(val);
      }
    }
  }
}

// ---------------------------------------------------------------------------
// Flash attention: D=32 heads, diagonal-masked softmax, online normalization.
// Grid: (N/64, B*H). Block: 128 threads = 4 waves, each wave: 16 query rows.
// QKV buffer layout: [b, n, {q|k|v}*C + h*32 + d], row stride 768, bf16.
// V tiles are streamed into LDS with double-buffered async copies
// (global_load_async_to_lds_b128, ASYNCcnt) so the next tile's HBM/L2 fetch
// overlaps the current tile's WMMA + softmax work.
// ---------------------------------------------------------------------------
__launch_bounds__(128)
__global__ void attn_kernel(const bf16_t* __restrict__ qkv,
                            const float* __restrict__ scale_ptr,
                            bf16_t* __restrict__ out) {
  const int N = 2048, C = 256, LDQ = 768, D = 32;
  __shared__ bf16_t Vbuf[2][32][40];  // row-major V tiles [key][d], 80B row stride
  __shared__ bf16_t PB[4][16][40];    // per-wave P tile: PB[w][m][k]

  int w = threadIdx.x >> 5, lane = threadIdx.x & 31;
  int g = lane >> 4, r = lane & 15;
  int bh = blockIdx.y;
  int b = bh >> 3, h = bh & 7;
  int q0 = blockIdx.x * 64 + w * 16;
  float scale = scale_ptr[0];

  const bf16_t* Qb = qkv + ((size_t)b * N) * LDQ + h * D;
  const bf16_t* Kb = Qb + 256;
  const bf16_t* Vb = Qb + 512;

  // Q fragment for this wave's 16 rows (K = full D = 32)
  v16bf qf = load_frag_a(Qb + (size_t)q0 * LDQ, LDQ);

  // Each of the 128 threads async-copies one 16B chunk of the 32x32 V tile.
  int vrow = threadIdx.x & 31;   // key within tile
  int vseg = threadIdx.x >> 5;   // 8-element d segment
  auto stage_v = [&](int kb, int sel) {
    unsigned lds_off = (unsigned)(size_t)&Vbuf[sel][vrow][vseg * 8];
    const bf16_t* gp = Vb + (size_t)(kb + vrow) * LDQ + vseg * 8;
    asm volatile("global_load_async_to_lds_b128 %0, %1, off"
                 :: "v"(lds_off), "v"(gp) : "memory");
  };

  stage_v(0, 0);
  asm volatile("s_wait_asynccnt 0x0" ::: "memory");
  __syncthreads();

  v8f o0{}, o1{};
  float mrow[8], lrow[8];
  #pragma unroll
  for (int j = 0; j < 8; ++j) { mrow[j] = -INFINITY; lrow[j] = 0.f; }

  int buf = 0;
  for (int kb = 0; kb < N; kb += 32) {
    // prefetch next V tile into the other buffer (overlaps all compute below)
    if (kb + 32 < N) stage_v(kb + 32, buf ^ 1);

    // S = Q * K^T for 32 keys (two 16-key halves), K operands direct from global
    v16bf kf0 = load_frag_b(Kb + (size_t)kb * LDQ, LDQ);
    v16bf kf1 = load_frag_b(Kb + (size_t)(kb + 16) * LDQ, LDQ);
    v8f s0{}, s1{};
    s0 = WMMA_BF16(qf, kf0, s0);
    s1 = WMMA_BF16(qf, kf1, s1);

    // online softmax (rows live across 16-lane groups; element j -> row 8g+j, col r)
    float p0[8], p1[8];
    #pragma unroll
    for (int j = 0; j < 8; ++j) {
      int row = q0 + 8 * g + j;
      float v0 = s0[j] * scale;
      float v1 = s1[j] * scale;
      if (kb + r == row)      v0 = -INFINITY;   // mask diagonal
      if (kb + 16 + r == row) v1 = -INFINITY;
      float mx = fmaxf(v0, v1);
      #pragma unroll
      for (int m = 1; m < 16; m <<= 1) mx = fmaxf(mx, __shfl_xor(mx, m, 32));
      float mn = fmaxf(mrow[j], mx);
      float al = __expf(mrow[j] - mn);
      float e0 = __expf(v0 - mn);
      float e1 = __expf(v1 - mn);
      float sum = e0 + e1;
      #pragma unroll
      for (int m = 1; m < 16; m <<= 1) sum += __shfl_xor(sum, m, 32);
      lrow[j] = lrow[j] * al + sum;
      mrow[j] = mn;
      o0[j] *= al;
      o1[j] *= al;
      p0[j] = e0;
      p1[j] = e1;
    }

    // transpose P (D-layout -> A-layout) through per-wave LDS tile
    #pragma unroll
    for (int j = 0; j < 8; ++j) {
      PB[w][8 * g + j][r]      = f2bf(p0[j]);
      PB[w][8 * g + j][16 + r] = f2bf(p1[j]);
    }

    // A fragment of P: lane r = row r, K = g*16 + e (contiguous)
    v8bf plo = *(const v8bf*)(&PB[w][r][g * 16]);
    v8bf phi = *(const v8bf*)(&PB[w][r][g * 16 + 8]);
    v16bf pf = cat16(plo, phi);

    // B fragments of V from the row-major async-staged tile:
    // lane r = output dim column, K (key) = g*16+e -> strided column gather
    v16bf vf0, vf1;
    #pragma unroll
    for (int e = 0; e < 16; ++e) {
      vf0[e] = Vbuf[buf][g * 16 + e][r];
      vf1[e] = Vbuf[buf][g * 16 + e][16 + r];
    }

    o0 = WMMA_BF16(pf, vf0, o0);
    o1 = WMMA_BF16(pf, vf1, o1);

    // next tile fully in LDS + everyone done reading this tile -> swap
    asm volatile("s_wait_asynccnt 0x0" ::: "memory");
    __syncthreads();
    buf ^= 1;
  }

  // normalize and write out (B, N, C) bf16, head offset h*32
  #pragma unroll
  for (int j = 0; j < 8; ++j) {
    int row = q0 + 8 * g + j;
    float inv = 1.0f / lrow[j];
    size_t base = ((size_t)b * N + row) * C + h * D;
    out[base + r]      = f2bf(o0[j] * inv);
    out[base + 16 + r] = f2bf(o1[j] * inv);
  }
}

// ---------------------------------------------------------------------------
// Host launcher
// ---------------------------------------------------------------------------
extern "C" void kernel_launch(void* const* d_in, const int* in_sizes, int n_in,
                              void* d_out, int out_size, void* d_ws, size_t ws_size,
                              hipStream_t stream) {
  const float* x      = (const float*)d_in[0];
  const float* ln1_g  = (const float*)d_in[1];
  const float* ln1_b  = (const float*)d_in[2];
  const float* qkv_w  = (const float*)d_in[3];
  const float* scale  = (const float*)d_in[4];
  const float* proj_w = (const float*)d_in[5];
  const float* proj_b = (const float*)d_in[6];
  const float* ln2_g  = (const float*)d_in[7];
  const float* ln2_b  = (const float*)d_in[8];
  const float* fc1_w  = (const float*)d_in[9];
  const float* fc1_b  = (const float*)d_in[10];
  const float* fc2_w  = (const float*)d_in[11];
  const float* fc2_b  = (const float*)d_in[12];
  float* out = (float*)d_out;
  (void)in_sizes; (void)n_in; (void)out_size; (void)ws_size;

  const int B = 8, N = 2048, C = 256, HID = 1024;
  const int M = B * N;  // 16384

  char* ws = (char*)d_ws;
  size_t off = 0;
  auto alloc = [&](size_t bytes) -> void* {
    void* p = ws + off;
    off += bytes;
    off = (off + 255) & ~(size_t)255;
    return p;
  };

  bf16_t* xn    = (bf16_t*)alloc((size_t)M * C * 2);
  bf16_t* wqkv  = (bf16_t*)alloc((size_t)3 * C * C * 2);
  bf16_t* wproj = (bf16_t*)alloc((size_t)C * C * 2);
  bf16_t* wfc1  = (bf16_t*)alloc((size_t)HID * C * 2);
  bf16_t* wfc2  = (bf16_t*)alloc((size_t)C * HID * 2);
  bf16_t* qkvb  = (bf16_t*)alloc((size_t)M * 3 * C * 2);
  bf16_t* attnb = (bf16_t*)alloc((size_t)M * C * 2);
  float*  x1    = (float*)alloc((size_t)M * C * 4);
  bf16_t* xn2   = (bf16_t*)alloc((size_t)M * C * 2);
  bf16_t* hbuf  = (bf16_t*)alloc((size_t)M * HID * 2);

  // 1) weights -> bf16
  cast_kernel<<<(3 * C * C + 255) / 256, 256, 0, stream>>>(qkv_w, wqkv, 3 * C * C);
  cast_kernel<<<(C * C + 255) / 256, 256, 0, stream>>>(proj_w, wproj, C * C);
  cast_kernel<<<(HID * C + 255) / 256, 256, 0, stream>>>(fc1_w, wfc1, HID * C);
  cast_kernel<<<(C * HID + 255) / 256, 256, 0, stream>>>(fc2_w, wfc2, C * HID);

  // 2) LN1
  ln_kernel<<<M, 256, 0, stream>>>(x, ln1_g, ln1_b, xn);

  // 3) QKV projection: (M,256) x (768,256)^T -> (M,768) bf16
  gemm_wmma_kernel<<<dim3(M / 128, (3 * C) / 64), 256, 0, stream>>>(
      xn, wqkv, C, 3 * C, nullptr, nullptr, qkvb, nullptr, 0);

  // 4) attention
  attn_kernel<<<dim3(N / 64, B * 8), 128, 0, stream>>>(qkvb, scale, attnb);

  // 5) output projection + bias + residual(x) -> x1 (fp32)
  gemm_wmma_kernel<<<dim3(M / 128, C / 64), 256, 0, stream>>>(
      attnb, wproj, C, C, proj_b, x, nullptr, x1, 0);

  // 6) LN2
  ln_kernel<<<M, 256, 0, stream>>>(x1, ln2_g, ln2_b, xn2);

  // 7) FC1 + bias + exact GELU -> hbuf (bf16)
  gemm_wmma_kernel<<<dim3(M / 128, HID / 64), 256, 0, stream>>>(
      xn2, wfc1, C, HID, fc1_b, nullptr, hbuf, nullptr, 1);

  // 8) FC2 + bias + residual(x1) -> out (fp32)
  gemm_wmma_kernel<<<dim3(M / 128, C / 64), 256, 0, stream>>>(
      hbuf, wfc2, HID, C, fc2_b, x1, nullptr, out, 0);
}
